// MultiHeadAttention_15169824489569
// MI455X (gfx1250) — matrix operand order repaired and checked by reading, not compiled, not run on hardware
//
#include <hip/hip_runtime.h>

typedef __bf16 bf16_t;
typedef __attribute__((ext_vector_type(16))) __bf16 v16bf;
typedef __attribute__((ext_vector_type(8)))  __bf16 v8bf;
typedef __attribute__((ext_vector_type(8)))  float  v8f;
typedef int v4i_vs __attribute__((vector_size(4 * sizeof(int))));

#define D_MODEL  1024
#define N_HEAD   16
#define HEAD_DIM 64
#define SEQ      2048
#define BATCH    4
#define MTOT     (BATCH * SEQ)   // 8192 tokens

#if __has_builtin(__builtin_amdgcn_global_load_async_to_lds_b128) && \
    __has_builtin(__builtin_amdgcn_s_wait_asynccnt)
#define USE_ASYNC_LDS 1
#else
#define USE_ASYNC_LDS 0
#endif

// Copy 16 bytes per lane from global to LDS (async on CDNA5, sync fallback).
static __device__ __forceinline__ void copy16_g2l(const bf16_t* g, bf16_t* l) {
#if USE_ASYNC_LDS
  __builtin_amdgcn_global_load_async_to_lds_b128(
      (__attribute__((address_space(1))) v4i_vs*)g,
      (__attribute__((address_space(3))) v4i_vs*)l, 0, 0);
#else
  *(v8bf*)l = *(const v8bf*)g;
#endif
}

static __device__ __forceinline__ void wait_async_all() {
#if USE_ASYNC_LDS
  __builtin_amdgcn_s_wait_asynccnt(0);
#endif
}

static __device__ __forceinline__ v8f vzero8() {
  v8f z;
#pragma unroll
  for (int i = 0; i < 8; ++i) z[i] = 0.0f;
  return z;
}

// A-fragment (16x32 bf16): lane (m = lane&15, half=lane>>4)
// holds K = half*16..+16 contiguous (matches B fragment loads).
static __device__ __forceinline__ v16bf make_afrag(const bf16_t* row, int half) {
  v8bf lo = *(const v8bf*)(row + half * 16);
  v8bf hi = *(const v8bf*)(row + half * 16 + 8);
  v16bf a;
#pragma unroll
  for (int e = 0; e < 8; ++e) { a[e] = lo[e]; a[8 + e] = hi[e]; }
  return a;
}

__global__ void cvt_f32_to_bf16(const float* __restrict__ in,
                                bf16_t* __restrict__ out, int n) {
  int i = blockIdx.x * blockDim.x + threadIdx.x;
  if (i < n) out[i] = (bf16_t)in[i];
}

// Wt[o][i] = W[i][o]  (WMMA B fragments read K-contiguous rows)
__global__ void cvt_transpose_w(const float* __restrict__ W,
                                bf16_t* __restrict__ Wt) {
  int idx = blockIdx.x * blockDim.x + threadIdx.x;   // idx = o*D + i
  int o = idx >> 10;
  int i = idx & (D_MODEL - 1);
  Wt[idx] = (bf16_t)W[i * D_MODEL + o];
}

// Vt[((b*16+h)*64+d)][t] = V[(b*2048+t)][h*64+d]
__global__ void transpose_v_heads(const bf16_t* __restrict__ V,
                                  bf16_t* __restrict__ Vt) {
  int idx = blockIdx.x * blockDim.x + threadIdx.x;
  int t = idx & (SEQ - 1);
  int d = (idx >> 11) & (HEAD_DIM - 1);
  int h = (idx >> 17) & (N_HEAD - 1);
  int b = idx >> 21;
  Vt[idx] = V[(size_t)(b * SEQ + t) * D_MODEL + h * HEAD_DIM + d];
}

// C = A[M][K] * Bt[N][K]^T + bias.
// Block = 4 waves sharing a 64-row A slab staged into LDS via async copies
// (double-buffered); each wave owns a 64x32 output tile (8 WMMAs / K-step).
template <bool OUTF32>
__global__ __launch_bounds__(128) void gemm_bf16_wmma(
    const bf16_t* __restrict__ A, const bf16_t* __restrict__ Bt,
    const float* __restrict__ bias, void* __restrict__ Cp,
    int M, int N, int K) {
  __shared__ __align__(32) bf16_t abuf[2][64 * 32];   // 2 x 4KB A tiles
  const int tid  = threadIdx.x;
  const int lane = tid & 31;
  const int wave = tid >> 5;
  const int half = lane >> 4;
  const int mn   = lane & 15;
  const int m0 = blockIdx.x * 64;
  const int n0 = blockIdx.y * 128 + wave * 32;
  const bf16_t* Ab = A + (size_t)m0 * K;

  v8f acc[4][2];
#pragma unroll
  for (int i = 0; i < 4; ++i)
#pragma unroll
    for (int j = 0; j < 2; ++j) acc[i][j] = vzero8();

  // stage A tile [64 x 32] at column k0 into abuf[sel]: 256 16B chunks,
  // 2 per thread; chunk ch -> row = ch>>2, col8 = ch&3, lds elem = ch*8.
  auto stageA = [&](int k0, int sel) {
#pragma unroll
    for (int u = 0; u < 2; ++u) {
      const int ch = tid * 2 + u;
      copy16_g2l(Ab + (size_t)(ch >> 2) * K + k0 + (ch & 3) * 8,
                 &abuf[sel][ch * 8]);
    }
  };

  stageA(0, 0);
  int cur = 0;
  for (int k0 = 0; k0 < K; k0 += 32) {
    wait_async_all();
    __syncthreads();
    if (k0 + 32 < K) stageA(k0 + 32, cur ^ 1);

    const bf16_t* al = abuf[cur];
    v16bf af[4];
#pragma unroll
    for (int i = 0; i < 4; ++i)
      af[i] = make_afrag(al + (16 * i + mn) * 32, half);
    v16bf bfr[2];
#pragma unroll
    for (int j = 0; j < 2; ++j)
      bfr[j] = *(const v16bf*)(Bt + (size_t)(n0 + 16 * j + mn) * K + k0 + half * 16);
#pragma unroll
    for (int i = 0; i < 4; ++i)
#pragma unroll
      for (int j = 0; j < 2; ++j)
        acc[i][j] = __builtin_amdgcn_wmma_f32_16x16x32_bf16(
            false, af[i], false, bfr[j], (short)0, acc[i][j], false, false);
    cur ^= 1;
  }

#pragma unroll
  for (int i = 0; i < 4; ++i)
#pragma unroll
    for (int j = 0; j < 2; ++j) {
      const int col = n0 + 16 * j + mn;
      const float bb = bias ? bias[col] : 0.0f;
      const int r0 = m0 + 16 * i + 8 * half;
      if (OUTF32) {
        float* C = (float*)Cp;
#pragma unroll
        for (int r = 0; r < 8; ++r)
          C[(size_t)(r0 + r) * N + col] = acc[i][j][r] + bb;
      } else {
        bf16_t* C = (bf16_t*)Cp;
#pragma unroll
        for (int r = 0; r < 8; ++r)
          C[(size_t)(r0 + r) * N + col] = (bf16_t)(acc[i][j][r] + bb);
      }
    }
}

// Causal flash attention. Block = one (batch, head) x 64 queries; 4 waves
// handle 16-query tiles and share async-LDS-staged K/V tiles (32 keys).
__global__ __launch_bounds__(128) void flash_attn_wmma(
    const bf16_t* __restrict__ Q, const bf16_t* __restrict__ K,
    const bf16_t* __restrict__ Vt, bf16_t* __restrict__ O) {
  __shared__ __align__(32) bf16_t kbuf[2][32 * 64];   // [key][d]
  __shared__ __align__(32) bf16_t vbuf[2][64 * 32];   // [d][key]
  __shared__ __align__(32) bf16_t plds[4][16 * 32];   // per-wave P tile
  const int tid  = threadIdx.x;
  const int lane = tid & 31;
  const int wave = tid >> 5;
  const int half = lane >> 4;
  const int mn   = lane & 15;

  const int qc = blockIdx.x & 31;          // 64-query chunk
  const int h  = (blockIdx.x >> 5) & 15;
  const int b  = blockIdx.x >> 9;
  const int q0 = qc * 64 + wave * 16;      // this wave's query tile

  const bf16_t* Kb = K + (size_t)(b * SEQ) * D_MODEL + h * HEAD_DIM;
  const bf16_t* Vb = Vt + (size_t)((b * N_HEAD + h) * HEAD_DIM) * SEQ;

  // stage K tile (32 keys x 64 d, row 128B = 8 chunks) and V tile
  // (64 d x 32 keys, row 64B = 4 chunks); 2 chunks each per thread.
  auto stageKV = [&](int kb, int sel) {
#pragma unroll
    for (int u = 0; u < 2; ++u) {
      const int ch = tid * 2 + u;          // 0..255
      copy16_g2l(Kb + (size_t)(kb + (ch >> 3)) * D_MODEL + (ch & 7) * 8,
                 &kbuf[sel][ch * 8]);
      copy16_g2l(Vb + (size_t)(ch >> 2) * SEQ + kb + (ch & 3) * 8,
                 &vbuf[sel][ch * 8]);
    }
  };

  // Q fragments (direct from global, used every iteration)
  const bf16_t* qrow = Q + (size_t)(b * SEQ + q0 + mn) * D_MODEL + h * HEAD_DIM;
  v16bf aq[2];
#pragma unroll
  for (int c = 0; c < 2; ++c) aq[c] = make_afrag(qrow + 32 * c, half);

  v8f o0 = vzero8(), o1 = vzero8(), o2 = vzero8(), o3 = vzero8();
  float mr[8], lr[8];
#pragma unroll
  for (int r = 0; r < 8; ++r) { mr[r] = -1e30f; lr[r] = 0.0f; }

  bf16_t* pl = plds[wave];
  const float sc = 0.125f;                 // 1/sqrt(64)
  const int kb_end = qc * 64 + 64;         // whole block's causal range

  stageKV(0, 0);
  int cur = 0;
  for (int kb = 0; kb < kb_end; kb += 32) {
    wait_async_all();
    __syncthreads();
    if (kb + 32 < kb_end) stageKV(kb + 32, cur ^ 1);

    if (kb < q0 + 16) {                    // wave-uniform causal guard
      const bf16_t* kl = kbuf[cur];
      const bf16_t* vl = vbuf[cur];

      // S(16x32) = Q(16x64) @ K^T : 4 WMMAs
      v8f s0 = vzero8(), s1 = vzero8();
#pragma unroll
      for (int c = 0; c < 2; ++c) {
        v16bf bk0 = *(const v16bf*)(kl + (mn)      * 64 + 32 * c + half * 16);
        v16bf bk1 = *(const v16bf*)(kl + (16 + mn) * 64 + 32 * c + half * 16);
        s0 = __builtin_amdgcn_wmma_f32_16x16x32_bf16(false, aq[c], false, bk0, (short)0, s0, false, false);
        s1 = __builtin_amdgcn_wmma_f32_16x16x32_bf16(false, aq[c], false, bk1, (short)0, s1, false, false);
      }

      const bool needmask = (kb + 31 > q0);
      float p0[8], p1[8];
#pragma unroll
      for (int r = 0; r < 8; ++r) {
        const int qg = q0 + 8 * half + r;
        float v0 = s0[r] * sc;
        float v1 = s1[r] * sc;
        if (needmask) {
          if (kb + mn > qg)      v0 = -1e30f;
          if (kb + 16 + mn > qg) v1 = -1e30f;
        }
        float mx = fmaxf(v0, v1);
        mx = fmaxf(mx, __shfl_xor(mx, 1, 32));
        mx = fmaxf(mx, __shfl_xor(mx, 2, 32));
        mx = fmaxf(mx, __shfl_xor(mx, 4, 32));
        mx = fmaxf(mx, __shfl_xor(mx, 8, 32));
        const float mnew = fmaxf(mr[r], mx);
        const float scl  = __expf(mr[r] - mnew);
        const float e0   = __expf(v0 - mnew);
        const float e1   = __expf(v1 - mnew);
        float rs = e0 + e1;
        rs += __shfl_xor(rs, 1, 32);
        rs += __shfl_xor(rs, 2, 32);
        rs += __shfl_xor(rs, 4, 32);
        rs += __shfl_xor(rs, 8, 32);
        lr[r] = lr[r] * scl + rs;
        mr[r] = mnew;
        o0[r] *= scl; o1[r] *= scl; o2[r] *= scl; o3[r] *= scl;
        p0[r] = e0; p1[r] = e1;
      }

      // C-layout -> row-major P in wave-private LDS
#pragma unroll
      for (int r = 0; r < 8; ++r) {
        const int row = 8 * half + r;
        pl[row * 32 + mn]      = (bf16_t)p0[r];
        pl[row * 32 + 16 + mn] = (bf16_t)p1[r];
      }
      v16bf ap = make_afrag(pl + mn * 32, half);

      // O(16x64) += P(16x32) @ V(32x64) : 4 WMMAs
      v16bf bv0 = *(const v16bf*)(vl + (mn)      * 32 + half * 16);
      v16bf bv1 = *(const v16bf*)(vl + (16 + mn) * 32 + half * 16);
      v16bf bv2 = *(const v16bf*)(vl + (32 + mn) * 32 + half * 16);
      v16bf bv3 = *(const v16bf*)(vl + (48 + mn) * 32 + half * 16);
      o0 = __builtin_amdgcn_wmma_f32_16x16x32_bf16(false, ap, false, bv0, (short)0, o0, false, false);
      o1 = __builtin_amdgcn_wmma_f32_16x16x32_bf16(false, ap, false, bv1, (short)0, o1, false, false);
      o2 = __builtin_amdgcn_wmma_f32_16x16x32_bf16(false, ap, false, bv2, (short)0, o2, false, false);
      o3 = __builtin_amdgcn_wmma_f32_16x16x32_bf16(false, ap, false, bv3, (short)0, o3, false, false);
    }
    cur ^= 1;
  }

#pragma unroll
  for (int r = 0; r < 8; ++r) {
    const float inv = 1.0f / lr[r];
    const size_t base = (size_t)(b * SEQ + q0 + 8 * half + r) * D_MODEL + h * HEAD_DIM;
    O[base + mn]      = (bf16_t)(o0[r] * inv);
    O[base + 16 + mn] = (bf16_t)(o1[r] * inv);
    O[base + 32 + mn] = (bf16_t)(o2[r] * inv);
    O[base + 48 + mn] = (bf16_t)(o3[r] * inv);
  }
}

extern "C" void kernel_launch(void* const* d_in, const int* in_sizes, int n_in,
                              void* d_out, int out_size, void* d_ws, size_t ws_size,
                              hipStream_t stream) {
  const float* x  = (const float*)d_in[0];
  const float* Wq = (const float*)d_in[1];
  const float* bq = (const float*)d_in[2];
  const float* Wk = (const float*)d_in[3];
  const float* bk = (const float*)d_in[4];
  const float* Wv = (const float*)d_in[5];
  const float* bv = (const float*)d_in[6];
  const float* Wp = (const float*)d_in[7];
  const float* bp = (const float*)d_in[8];

  char* ws = (char*)d_ws;
  const size_t MB = 1024ull * 1024ull;
  bf16_t* xb  = (bf16_t*)(ws + 0 * MB);    // 16 MB x bf16
  bf16_t* wtq = (bf16_t*)(ws + 16 * MB);   // 2 MB each, W^T bf16
  bf16_t* wtk = (bf16_t*)(ws + 18 * MB);
  bf16_t* wtv = (bf16_t*)(ws + 20 * MB);
  bf16_t* wtp = (bf16_t*)(ws + 22 * MB);
  bf16_t* qb  = (bf16_t*)(ws + 24 * MB);   // 16 MB Q
  bf16_t* kbf = (bf16_t*)(ws + 40 * MB);   // 16 MB K
  bf16_t* vbf = (bf16_t*)(ws + 56 * MB);   // 16 MB V
  bf16_t* vtb = (bf16_t*)(ws + 72 * MB);   // 16 MB V^T per head
  bf16_t* ab  = (bf16_t*)(ws + 88 * MB);   // 16 MB attention output

  const int nx = MTOT * D_MODEL;           // 8388608
  const int nw = D_MODEL * D_MODEL;        // 1048576

  cvt_f32_to_bf16<<<nx / 256, 256, 0, stream>>>(x, xb, nx);
  cvt_transpose_w<<<nw / 256, 256, 0, stream>>>(Wq, wtq);
  cvt_transpose_w<<<nw / 256, 256, 0, stream>>>(Wk, wtk);
  cvt_transpose_w<<<nw / 256, 256, 0, stream>>>(Wv, wtv);
  cvt_transpose_w<<<nw / 256, 256, 0, stream>>>(Wp, wtp);

  dim3 gg(MTOT / 64, D_MODEL / 128);       // (128, 8)
  gemm_bf16_wmma<false><<<gg, 128, 0, stream>>>(xb, wtq, bq, qb, MTOT, D_MODEL, D_MODEL);
  gemm_bf16_wmma<false><<<gg, 128, 0, stream>>>(xb, wtk, bk, kbf, MTOT, D_MODEL, D_MODEL);
  gemm_bf16_wmma<false><<<gg, 128, 0, stream>>>(xb, wtv, bv, vbf, MTOT, D_MODEL, D_MODEL);

  transpose_v_heads<<<nx / 256, 256, 0, stream>>>(vbf, vtb);

  // 4*16*(2048/64) = 2048 blocks, each: one (b,h), 64 queries, 4 waves
  flash_attn_wmma<<<BATCH * N_HEAD * (SEQ / 64), 128, 0, stream>>>(qb, kbf, vtb, ab);

  gemm_bf16_wmma<true><<<gg, 128, 0, stream>>>(ab, wtp, bp, (float*)d_out, MTOT, D_MODEL, D_MODEL);
}
